// Attention_70463233458510
// MI455X (gfx1250) — compile-verified, matrix operand order checked by reading, not run on hardware
//
#include <hip/hip_runtime.h>

#define BB 2
#define LL 4096
#define HH 8
#define DD 64
#define MM 256
#define CHUNK 64
#define NCH (LL / CHUNK)   // 64 chunks
#define BH (BB * HH)       // 16
#define EPSV 0.001f
#define RATIO 0.0625f      // 1/sqrt(M)

typedef __attribute__((ext_vector_type(16))) _Float16 v16h;
typedef __attribute__((ext_vector_type(8)))  _Float16 h8;
typedef __attribute__((ext_vector_type(8)))  float    v8f;

// Load a 16x32 f16 WMMA A-fragment (or B^T rows used as the B operand) from a
// row-major buffer with row stride `ld` (elements), starting at column kbase.
// Layout per CDNA5 ISA 7.12.2: lanes 0-15 -> row=lane, K = kbase+{0..7, 16..23};
// lanes 16-31 -> row=lane-16, K = kbase+{8..15, 24..31}.
__device__ __forceinline__ v16h load_frag(const _Float16* base, int ld, int kbase) {
  const int lane  = threadIdx.x & 31;
  const int row   = lane & 15;
  const int khalf = (lane >> 4) & 1;
  const _Float16* p = base + row * ld + kbase + khalf * 8;
  union { v16h v; h8 h[2]; } u;
  u.h[0] = *(const h8*)(p);
  u.h[1] = *(const h8*)(p + 16);
  return u.v;
}

__device__ __forceinline__ v8f wmma_f16(v16h a, v16h b, v8f c) {
  // D = A(16x32) * B(32x16) + C, f32 accumulate
  return __builtin_amdgcn_wmma_f32_16x16x32_f16(false, a, false, b, (short)0, c,
                                                false, false);
}

// --- gfx1250 async global->LDS DMA (ASYNCcnt-tracked) ----------------------
// Generic pointer to a __shared__ object: low 32 bits are the LDS byte offset
// (flat LDS aperture maps addr[31:0] -> LDS address).
__device__ __forceinline__ unsigned lds_off_of(const void* p) {
  return (unsigned)(size_t)p;
}
__device__ __forceinline__ void async_b128(unsigned lds, const void* g) {
  asm volatile("global_load_async_to_lds_b128 %0, %1, off"
               :: "v"(lds), "v"(g) : "memory");
}
__device__ __forceinline__ void async_b32(unsigned lds, const void* g) {
  asm volatile("global_load_async_to_lds_b32 %0, %1, off"
               :: "v"(lds), "v"(g) : "memory");
}
__device__ __forceinline__ void wait_async() {
  asm volatile("s_wait_asynccnt 0" ::: "memory");
}

// ---------------------------------------------------------------------------
// Kernel 1: rotary embedding + random-feature projection (relu kernel).
// Block: 256 threads (8 waves). Grid: (BH*NCH, 2) ; y=0 -> q, y=1 -> k.
// Out tile per block: 64 rows x 256 features, K = D = 64.
// qp/kp layout: [bh][l][m] (rows directly usable as WMMA fragments).
// ---------------------------------------------------------------------------
__global__ void __launch_bounds__(256) k_rope_project(
    const float* __restrict__ q, const float* __restrict__ k,
    const float* __restrict__ sinu, const float* __restrict__ proj,
    _Float16* __restrict__ qp, _Float16* __restrict__ kp) {
  __shared__ __align__(16) _Float16 Arot[CHUNK][72];
  __shared__ __align__(16) _Float16 Pj[MM][72];
  const int tid = threadIdx.x;
  const int bh  = blockIdx.x / NCH;
  const int ct  = blockIdx.x % NCH;
  const int b = bh / HH, h = bh % HH;
  const int l0 = ct * CHUNK;
  const float* src = (blockIdx.y == 0) ? q : k;
  _Float16*    dst = (blockIdx.y == 0) ? qp : kp;

  __builtin_prefetch(proj + tid * 16, 0, 0);  // global_prefetch: proj is hot in L2

  // stage projection matrix as f16 (rows m, contiguous d) -- coalesced reads
  for (int p = tid; p < MM * DD; p += 256) {
    int d = p & (DD - 1), m = p >> 6;
    Pj[m][d] = (_Float16)proj[(size_t)m * DD + d];
  }
  // rotate rows into LDS: qrot[2i] = x0*cos - x1*sin ; qrot[2i+1] = x1*cos + x0*sin
  for (int p = tid; p < CHUNK * (DD / 2); p += 256) {
    int row = p >> 5, pr = p & 31;
    int l = l0 + row;
    size_t sidx = (((size_t)b * LL + l) * HH + h) * DD + 2 * pr;
    float x0 = src[sidx];
    float x1 = src[sidx + 1];
    float sn = sinu[l * DD + pr];
    float cs = sinu[l * DD + 32 + pr];
    Arot[row][2 * pr]     = (_Float16)(x0 * cs - x1 * sn);
    Arot[row][2 * pr + 1] = (_Float16)(x1 * cs + x0 * sn);
  }
  __syncthreads();

  const int wave = tid >> 5;
  const int lane = tid & 31;
  const int rt = wave >> 1;      // row tile 0..3
  const int chalf = wave & 1;    // column half
  for (int c8 = 0; c8 < 8; ++c8) {
    int mt = chalf * 8 + c8;     // feature tile 0..15
    v8f acc = {};
    for (int ks = 0; ks < 2; ++ks) {
      v16h a  = load_frag(&Arot[rt * 16][0], 72, ks * 32);
      v16h bf = load_frag(&Pj[mt * 16][0], 72, ks * 32);
      acc = wmma_f16(a, bf, acc);
    }
    for (int r = 0; r < 8; ++r) {
      int row = rt * 16 + r + ((lane >> 4) ? 8 : 0);
      int col = mt * 16 + (lane & 15);
      float val = fmaxf(acc[r] * RATIO, 0.f) + EPSV;
      dst[((size_t)bh * LL + (l0 + row)) * MM + col] = (_Float16)val;
    }
  }
}

// ---------------------------------------------------------------------------
// Kernel 1b: v (B,L,H,D) fp32 -> vt [bh][d][l] f16 (tiled transpose via LDS,
// coalesced on both sides). Grid: BH*NCH blocks.
// ---------------------------------------------------------------------------
__global__ void __launch_bounds__(256) k_vtrans(const float* __restrict__ v,
                                                _Float16* __restrict__ vt) {
  __shared__ _Float16 T[DD][72];
  const int tid = threadIdx.x;
  const int bh = blockIdx.x / NCH;
  const int ct = blockIdx.x % NCH;
  const int b = bh / HH, h = bh % HH;
  const int l0 = ct * CHUNK;
  for (int p = tid; p < CHUNK * DD; p += 256) {
    int d = p & 63, t = p >> 6;
    T[d][t] = (_Float16)v[(((size_t)b * LL + (l0 + t)) * HH + h) * DD + d];
  }
  __syncthreads();
  for (int p = tid; p < CHUNK * DD; p += 256) {
    int t = p & 63, d = p >> 6;
    vt[((size_t)bh * DD + d) * LL + l0 + t] = T[d][t];
  }
}

// ---------------------------------------------------------------------------
// Kernel 2: per-chunk KV_c = kp_c^T (M x T) @ v_c (T x D) and ksum_c.
// ckv layout: [bh][c][d][m] (transposed, ready for kernel 4's B operand).
// Grid: BH*NCH blocks, 256 threads.
// ---------------------------------------------------------------------------
__global__ void __launch_bounds__(256) k_chunk_sums(
    const _Float16* __restrict__ kp, const _Float16* __restrict__ vt,
    _Float16* __restrict__ ckv, float* __restrict__ cks) {
  __shared__ __align__(16) _Float16 KpT[MM][72];     // [m][t]
  __shared__ __align__(16) _Float16 VT[CHUNK][72];   // [d][t]
  const int tid = threadIdx.x;
  const int bh = blockIdx.x / NCH;
  const int c  = blockIdx.x % NCH;
  const int l0 = c * CHUNK;
  // v tile: straight rows from vt[bh][d][l] -> async DMA to LDS
  const _Float16* vtb = vt + (size_t)bh * DD * LL + l0;
  for (int p = tid; p < CHUNK * 8; p += 256) {       // 512 x 16B
    int d = p >> 3, e = p & 7;
    async_b128(lds_off_of(&VT[d][e * 8]), vtb + (size_t)d * LL + e * 8);
  }
  // kp^T tile: transpose in LDS (rows m, contiguous t: this kernel's A operand)
  for (int p = tid; p < MM * CHUNK; p += 256) {
    int m = p & (MM - 1), t = p >> 8;
    KpT[m][t] = kp[((size_t)bh * LL + l0 + t) * MM + m];
  }
  wait_async();
  __syncthreads();
  const int wave = tid >> 5, lane = tid & 31;
  for (int mt2 = 0; mt2 < 2; ++mt2) {
    int mt = wave * 2 + mt2;                 // 16 feature tiles
    for (int dt = 0; dt < 4; ++dt) {
      v8f acc = {};
      for (int ks = 0; ks < 2; ++ks) {
        v16h a  = load_frag(&KpT[mt * 16][0], 72, ks * 32);
        v16h bf = load_frag(&VT[dt * 16][0], 72, ks * 32);
        acc = wmma_f16(a, bf, acc);
      }
      for (int r = 0; r < 8; ++r) {
        int m = mt * 16 + r + ((lane >> 4) ? 8 : 0);
        int d = dt * 16 + (lane & 15);
        ckv[(((size_t)bh * NCH + c) * DD + d) * MM + m] = (_Float16)acc[r];
      }
    }
  }
  float s = 0.f;
  for (int t = 0; t < CHUNK; ++t) s += (float)KpT[tid][t];
  cks[((size_t)bh * NCH + c) * MM + tid] = s;
}

// ---------------------------------------------------------------------------
// Kernel 3: exclusive prefix over chunks (fp32 running sums, f16 storage).
// Both ckv and pkv are [bh][c][d][m]; flat element scan is coalesced.
// Grid: BH blocks, 256 threads.
// ---------------------------------------------------------------------------
__global__ void __launch_bounds__(256) k_prefix(
    const _Float16* __restrict__ ckv, const float* __restrict__ cks,
    _Float16* __restrict__ pkv, float* __restrict__ pks) {
  const int tid = threadIdx.x;
  const int bh = blockIdx.x;
  for (int e = tid; e < MM * DD; e += 256) {
    float run = 0.f;
    for (int c = 0; c < NCH; ++c) {
      size_t idx = ((size_t)bh * NCH + c) * (MM * DD) + e;
      pkv[idx] = (_Float16)run;
      run += (float)ckv[idx];
    }
  }
  float run = 0.f;
  for (int c = 0; c < NCH; ++c) {
    size_t idx = ((size_t)bh * NCH + c) * MM + tid;
    pks[idx] = run;
    run += cks[idx];
  }
}

// ---------------------------------------------------------------------------
// Kernel 4: per-chunk output.
//   S = mask(qp_c @ kp_c^T)           (64x64, K=256)
//   den[t] = qp[t].ksum_pref + rowsum(S[t])
//   num = qp_c @ KV_pref + S @ v_c    (64x64, K=256 + K=64, same accumulators)
//   out = num / den
// All LDS staging is straight-row async DMA. Grid: BH*NCH, 256 threads.
// ---------------------------------------------------------------------------
__global__ void __launch_bounds__(256) k_attn_chunk(
    const _Float16* __restrict__ qp, const _Float16* __restrict__ kp,
    const _Float16* __restrict__ vt, const _Float16* __restrict__ pkv,
    const float* __restrict__ pks, float* __restrict__ out) {
  __shared__ __align__(16) _Float16 VT[CHUNK][72];    // [d][t]
  __shared__ __align__(16) _Float16 KVT[DD][264];     // [d][m]
  __shared__ __align__(16) _Float16 Sh[CHUNK][72];    // [t][s], masked
  __shared__ __align__(16) float ksP[MM];
  __shared__ float den[CHUNK];
  const int tid = threadIdx.x;
  const int bh = blockIdx.x / NCH;
  const int c  = blockIdx.x % NCH;
  const int b = bh / HH, h = bh % HH;
  const int l0 = c * CHUNK;

  const _Float16* vtb = vt + (size_t)bh * DD * LL + l0;
  const _Float16* kvb = pkv + ((size_t)bh * NCH + c) * (MM * DD);  // [d][m]
  for (int p = tid; p < CHUNK * 8; p += 256) {        // v tile, 512 x 16B
    int d = p >> 3, e = p & 7;
    async_b128(lds_off_of(&VT[d][e * 8]), vtb + (size_t)d * LL + e * 8);
  }
  for (int p = tid; p < DD * 32; p += 256) {          // KV prefix, 2048 x 16B
    int d = p >> 5, e = p & 31;
    async_b128(lds_off_of(&KVT[d][e * 8]), kvb + (size_t)d * MM + e * 8);
  }
  async_b32(lds_off_of(&ksP[tid]), pks + ((size_t)bh * NCH + c) * MM + tid);
  if (tid < CHUNK) den[tid] = 0.f;
  wait_async();
  __syncthreads();

  const int wave = tid >> 5, lane = tid & 31;
  const int tt = wave >> 1;  // query-row tile 0..3
  const _Float16* qbase = qp + ((size_t)bh * LL + l0) * MM;
  const _Float16* kbase = kp + ((size_t)bh * LL + l0) * MM;

  // Phase 1: S tiles (each wave owns 2 of 16), causal mask, store f16 to LDS
  for (int st2 = 0; st2 < 2; ++st2) {
    int st = (wave & 1) * 2 + st2;
    v8f acc = {};
    for (int ks = 0; ks < 8; ++ks) {
      v16h a  = load_frag(qbase + (size_t)tt * 16 * MM, MM, ks * 32);
      v16h bf = load_frag(kbase + (size_t)st * 16 * MM, MM, ks * 32);
      acc = wmma_f16(a, bf, acc);
    }
    for (int r = 0; r < 8; ++r) {
      int t = tt * 16 + r + ((lane >> 4) ? 8 : 0);
      int s = st * 16 + (lane & 15);
      float val = (s <= t) ? acc[r] : 0.f;
      Sh[t][s] = (_Float16)val;
    }
  }
  __syncthreads();

  // Phase 2: denominators -- 4 partial threads per row, ds_add_f32 combine
  {
    int t = tid & 63, part = tid >> 6;
    float acc = 0.f;
    for (int s = part * 16; s < part * 16 + 16; ++s) acc += (float)Sh[t][s];
    const _Float16* qrow = qbase + (size_t)t * MM;
    for (int m = part * 64; m < part * 64 + 64; ++m)
      acc += (float)qrow[m] * ksP[m];
    atomicAdd(&den[t], acc);
  }
  __syncthreads();

  // Phase 3: numerators (inter + intra into same accumulators), divide, store
  for (int dt2 = 0; dt2 < 2; ++dt2) {
    int dt = (wave & 1) * 2 + dt2;
    v8f acc = {};
    for (int ks = 0; ks < 8; ++ks) {
      v16h a  = load_frag(qbase + (size_t)tt * 16 * MM, MM, ks * 32);
      v16h bf = load_frag(&KVT[dt * 16][0], 264, ks * 32);
      acc = wmma_f16(a, bf, acc);
    }
    for (int ks = 0; ks < 2; ++ks) {
      v16h a  = load_frag(&Sh[tt * 16][0], 72, ks * 32);
      v16h bf = load_frag(&VT[dt * 16][0], 72, ks * 32);
      acc = wmma_f16(a, bf, acc);
    }
    for (int r = 0; r < 8; ++r) {
      int t = tt * 16 + r + ((lane >> 4) ? 8 : 0);
      int d = dt * 16 + (lane & 15);
      int l = l0 + t;
      out[(((size_t)b * LL + l) * HH + h) * DD + d] = acc[r] / den[t];
    }
  }
}

// ---------------------------------------------------------------------------
extern "C" void kernel_launch(void* const* d_in, const int* in_sizes, int n_in,
                              void* d_out, int out_size, void* d_ws, size_t ws_size,
                              hipStream_t stream) {
  const float* q    = (const float*)d_in[0];
  const float* k    = (const float*)d_in[1];
  const float* v    = (const float*)d_in[2];
  const float* sinu = (const float*)d_in[3];
  const float* proj = (const float*)d_in[4];
  float* out = (float*)d_out;

  char* ws = (char*)d_ws;
  const size_t qp_e  = (size_t)BH * LL * MM;        // 16.8M elems
  const size_t vt_e  = (size_t)BH * LL * DD;        // 4.2M
  const size_t ckv_e = (size_t)BH * NCH * MM * DD;  // 16.8M
  const size_t cks_e = (size_t)BH * NCH * MM;       // 262k
  _Float16* qp  = (_Float16*)ws; ws += qp_e * 2;
  _Float16* kp  = (_Float16*)ws; ws += qp_e * 2;
  _Float16* vt  = (_Float16*)ws; ws += vt_e * 2;
  _Float16* ckv = (_Float16*)ws; ws += ckv_e * 2;
  _Float16* pkv = (_Float16*)ws; ws += ckv_e * 2;
  float*    cks = (float*)ws;    ws += cks_e * 4;
  float*    pks = (float*)ws;    ws += cks_e * 4;

  dim3 blk(256);
  k_rope_project<<<dim3(BH * NCH, 2), blk, 0, stream>>>(q, k, sinu, proj, qp, kp);
  k_vtrans<<<dim3(BH * NCH), blk, 0, stream>>>(v, vt);
  k_chunk_sums<<<dim3(BH * NCH), blk, 0, stream>>>(kp, vt, ckv, cks);
  k_prefix<<<dim3(BH), blk, 0, stream>>>(ckv, cks, pkv, pks);
  k_attn_chunk<<<dim3(BH * NCH), blk, 0, stream>>>(qp, kp, vt, pkv, pks, out);
}
